// ConvexReLU_83013127897747
// MI455X (gfx1250) — matrix operand order repaired and checked by reading, not compiled, not run on hardware
//
#include <hip/hip_runtime.h>

// ConvexReLU refactored:  out = x_flat @ F,  F = G^T-contracted (v - w)
//   Pass 1: F[k, l*128+m] = sum_j G[j,k] * (v-w)[j, l*128+m]   (1024 x 32768 GEMM, K=1024)
//   Pass 2: out[i,m]      = sum_kk x[i,kk] * F[kk, m]          (256 x 128 GEMM, K=262144, split-K + f32 atomics)
// v_wmma_f32_16x16x32_bf16 with f32 accumulate; f32 -> bf16 converted in pairs
// (packed cvt) on the way into LDS; LDS double-buffered; all WMMA fragments are
// two 16B-aligned ds_load_b128 per matrix.  F lives in d_ws as bf16 (64 MB).

typedef __attribute__((ext_vector_type(16))) __bf16    v16bf;
typedef __attribute__((ext_vector_type(2)))  __bf16    v2bf;
typedef __attribute__((ext_vector_type(8)))  float     v8f;
typedef __attribute__((ext_vector_type(2)))  float     v2f;
typedef __attribute__((ext_vector_type(4)))  unsigned  v4u;

#define BATCH   256
#define NEURONS 1024
#define NFEAT   1024
#define INDIM   256
#define OUTDIM  128
#define NC1     (INDIM * OUTDIM)   // 32768  : columns of F
#define K2      (NFEAT * INDIM)    // 262144 : pass-2 contraction depth
#define KSPLIT  64                 // pass-2 split-K workgroups per M-block
#define PITCH   40                 // LDS row pitch (ushorts): 80 B -> 16 B aligned rows

__device__ __forceinline__ unsigned short f32_bf16(float f) {
  unsigned u;
  __builtin_memcpy(&u, &f, 4);
  u += 0x7FFFu + ((u >> 16) & 1u);          // round-to-nearest-even
  return (unsigned short)(u >> 16);
}

// packed f32x2 -> bf16x2 (RTNE); lowers to v_cvt_pk_bf16_f32 when available
__device__ __forceinline__ unsigned pack2(float a, float b) {
  v2f f = {a, b};
  v2bf h = __builtin_convertvector(f, v2bf);
  unsigned u;
  __builtin_memcpy(&u, &h, 4);
  return u;
}

__global__ __launch_bounds__(256) void zero_out_kernel(float* __restrict__ out) {
  out[blockIdx.x * 256 + threadIdx.x] = 0.0f;
}

// ---------------------------------------------------------------------------
// Pass 1: F (1024 x 32768, bf16) = G^T (1024k x 1024j) @ D (1024j x 32768n)
// grid = (NC1/128, NFEAT/128), block = 256 (8 waves)
// ---------------------------------------------------------------------------
__global__ __launch_bounds__(256) void pass1_GtD_kernel(
    const float* __restrict__ G, const float* __restrict__ v,
    const float* __restrict__ w, unsigned short* __restrict__ F) {
  __shared__ __align__(16) unsigned short As[2][128][PITCH];  // [k-local][j-local]
  __shared__ __align__(16) unsigned short Bs[2][128][PITCH];  // [n-local][j-local]

  const int t    = threadIdx.x;
  const int nb   = blockIdx.x * 128;      // column block in [0, 32768)
  const int kb   = blockIdx.y * 128;      // row block in [0, 1024)
  const int wv   = t >> 5;                // wave 0..7 -> 16-row strip
  const int lane = t & 31;
  const int hf   = lane >> 4;             // lane half (wave32 WMMA layout)
  const int lr   = lane & 15;

  const v8f vzero = {0.f, 0.f, 0.f, 0.f, 0.f, 0.f, 0.f, 0.f};
  v8f acc[8];
#pragma unroll
  for (int nt = 0; nt < 8; ++nt) acc[nt] = vzero;

  // fill one (A,B) panel pair for contraction block jb into buffer `buf`
  auto fill = [&](int jb, int buf) {
#pragma unroll
    for (int it = 0; it < 8; ++it) {            // 2048 bf16-pairs, 256 threads
      int idx = t + 256 * it;
      int jl2 = idx >> 7, kl = idx & 127;       // pair of adjacent j, one k
      int j0  = jb + 2 * jl2;
      *(unsigned*)&As[buf][kl][2 * jl2] =
          pack2(G[(size_t)j0 * NFEAT + kb + kl],
                G[(size_t)(j0 + 1) * NFEAT + kb + kl]);
    }
#pragma unroll
    for (int it = 0; it < 8; ++it) {
      int idx = t + 256 * it;
      int jl2 = idx >> 7, nl = idx & 127;
      size_t o0 = (size_t)(jb + 2 * jl2) * NC1 + nb + nl;
      *(unsigned*)&Bs[buf][nl][2 * jl2] =
          pack2(v[o0] - w[o0], v[o0 + NC1] - w[o0 + NC1]);
    }
  };

  fill(0, 0);
  __syncthreads();

  for (int s = 0; s < NEURONS / 32; ++s) {
    const int buf = s & 1;
    if (s + 1 < NEURONS / 32) fill((s + 1) * 32, buf ^ 1);

    // A fragment: ushorts {0..7}+8hf and {16..23}+8hf of row -> two b128 loads
    v16bf a;
    v4u* a4 = (v4u*)&a;
    const v4u* arow = (const v4u*)&As[buf][16 * wv + lr][8 * hf];
    a4[0] = arow[0];
    a4[1] = arow[2];
#pragma unroll
    for (int nt = 0; nt < 8; ++nt) {
      // B fragment: ushorts {0..7}+8hf and {16..23}+8hf of row -> two b128 loads
      v16bf b;
      v4u* b4 = (v4u*)&b;
      const v4u* brow = (const v4u*)&Bs[buf][16 * nt + lr][8 * hf];
      b4[0] = brow[0];
      b4[1] = brow[2];
      acc[nt] = __builtin_amdgcn_wmma_f32_16x16x32_bf16(
          false, a, false, b, (short)0, acc[nt], false, false);
    }
    __syncthreads();
  }

  // C/D layout: VGPR r -> M = r + 8*hf, N = lr
#pragma unroll
  for (int nt = 0; nt < 8; ++nt)
#pragma unroll
    for (int r = 0; r < 8; ++r) {
      int row = kb + 16 * wv + 8 * hf + r;
      int col = nb + 16 * nt + lr;
      F[(size_t)row * NC1 + col] = f32_bf16(acc[nt][r]);
    }
}

// ---------------------------------------------------------------------------
// Pass 2: out (256 x 128, f32) += x_flat (256 x 262144) @ F (262144 x 128)
// grid = (KSPLIT, BATCH/128), block = 256; split-K reduced with f32 atomics
// ---------------------------------------------------------------------------
__global__ __launch_bounds__(256) void pass2_xF_kernel(
    const float* __restrict__ x, const unsigned short* __restrict__ F,
    float* __restrict__ out) {
  __shared__ __align__(16) unsigned short Xs[2][128][PITCH];  // [i-local][kk-local]
  __shared__ __align__(16) unsigned short Bs[2][128][PITCH];  // [m][kk-local]

  const int t    = threadIdx.x;
  const int ib   = blockIdx.y * 128;                 // batch-row block
  const int kk0  = blockIdx.x * (K2 / KSPLIT);       // 4096-wide K slab
  const int wv   = t >> 5;
  const int lane = t & 31;
  const int hf   = lane >> 4;
  const int lr   = lane & 15;

  const v8f vzero = {0.f, 0.f, 0.f, 0.f, 0.f, 0.f, 0.f, 0.f};
  v8f acc[8];
#pragma unroll
  for (int nt = 0; nt < 8; ++nt) acc[nt] = vzero;

  auto fill = [&](int kkb, int buf) {
#pragma unroll
    for (int it = 0; it < 8; ++it) {           // 128 rows x 16 kk-pairs
      int idx = t + 256 * it;
      int il = idx >> 4, kkl2 = idx & 15;      // x contiguous in kk -> b64 load
      v2f xv = *(const v2f*)&x[(size_t)(ib + il) * K2 + kkb + 2 * kkl2];
      *(unsigned*)&Xs[buf][il][2 * kkl2] = pack2(xv.x, xv.y);
    }
#pragma unroll
    for (int it = 0; it < 8; ++it) {           // 16 kk-pairs x 128 m
      int idx = t + 256 * it;
      int kkl2 = idx >> 7, nl = idx & 127;
      unsigned s0 = F[(size_t)(kkb + 2 * kkl2) * OUTDIM + nl];
      unsigned s1 = F[(size_t)(kkb + 2 * kkl2 + 1) * OUTDIM + nl];
      *(unsigned*)&Bs[buf][nl][2 * kkl2] = s0 | (s1 << 16);
    }
  };

  fill(kk0, 0);
  __syncthreads();

  const int steps = (K2 / KSPLIT) / 32;        // 128
  for (int s = 0; s < steps; ++s) {
    const int buf = s & 1;
    if (s + 1 < steps) fill(kk0 + (s + 1) * 32, buf ^ 1);

    v16bf a;
    v4u* a4 = (v4u*)&a;
    const v4u* arow = (const v4u*)&Xs[buf][16 * wv + lr][8 * hf];
    a4[0] = arow[0];
    a4[1] = arow[2];
#pragma unroll
    for (int nt = 0; nt < 8; ++nt) {
      v16bf b;
      v4u* b4 = (v4u*)&b;
      const v4u* brow = (const v4u*)&Bs[buf][16 * nt + lr][8 * hf];
      b4[0] = brow[0];
      b4[1] = brow[2];
      acc[nt] = __builtin_amdgcn_wmma_f32_16x16x32_bf16(
          false, a, false, b, (short)0, acc[nt], false, false);
    }
    __syncthreads();
  }

  // split-K reduction into d_out (zeroed beforehand) -> global_atomic_add_f32
#pragma unroll
  for (int nt = 0; nt < 8; ++nt)
#pragma unroll
    for (int r = 0; r < 8; ++r) {
      int row = ib + 16 * wv + 8 * hf + r;
      int col = 16 * nt + lr;
      atomicAdd(&out[(size_t)row * OUTDIM + col], acc[nt][r]);
    }
}

extern "C" void kernel_launch(void* const* d_in, const int* in_sizes, int n_in,
                              void* d_out, int out_size, void* d_ws, size_t ws_size,
                              hipStream_t stream) {
  const float* x = (const float*)d_in[0];   // (256, 1024, 256)
  const float* G = (const float*)d_in[1];   // (1024, 1024)
  const float* v = (const float*)d_in[2];   // (1024, 256, 128)
  const float* w = (const float*)d_in[3];   // (1024, 256, 128)
  float* out = (float*)d_out;               // (256, 128)
  unsigned short* F = (unsigned short*)d_ws;  // bf16 F: 1024*32768*2 = 64 MB

  // d_out is poisoned by the harness; pass 2 accumulates atomically -> zero it.
  zero_out_kernel<<<dim3((BATCH * OUTDIM) / 256), dim3(256), 0, stream>>>(out);

  pass1_GtD_kernel<<<dim3(NC1 / 128, NFEAT / 128), dim3(256), 0, stream>>>(G, v, w, F);

  pass2_xF_kernel<<<dim3(KSPLIT, BATCH / 128), dim3(256), 0, stream>>>(x, F, out);
}
